// BoundaryExtractionModule_34248069218322
// MI455X (gfx1250) — matrix operand order repaired and checked by reading, not compiled, not run on hardware
//
#include <hip/hip_runtime.h>
#include <hip/hip_bf16.h>

typedef _Float16 v16h __attribute__((ext_vector_type(16)));
typedef _Float16 v8h  __attribute__((ext_vector_type(8)));
typedef float    v8f  __attribute__((ext_vector_type(8)));

static __device__ __forceinline__ v8f wmma16x16x32_f16(v16h a, v16h b, v8f c) {
  // D = A(16x32 f16) * B(32x16 f16) + C(16x16 f32)
  return __builtin_amdgcn_wmma_f32_16x16x32_f16(
      /*neg_a=*/false, a, /*neg_b=*/false, b,
      /*c_mod=*/(short)0, c, /*reuse_a=*/false, /*reuse_b=*/false);
}

static __device__ __forceinline__ v16h combine_v8h(v8h lo, v8h hi) {
  return __builtin_shufflevector(lo, hi, 0, 1, 2, 3, 4, 5, 6, 7,
                                         8, 9, 10, 11, 12, 13, 14, 15);
}

// A-matrix 16x32 f16 fragment (ISA 7.12.2 layout):
// lanes 0-15: row M=lane, K={0..7} in v0-3, K={16..23} in v4-7
// lanes 16-31: row M=lane-16, K={8..15} and K={24..31}
static __device__ __forceinline__ v16h load_fragA(const _Float16* base,
                                                  int rowStride, int kb, int lane) {
  int row  = lane & 15;
  int koff = (lane >> 4) << 3;  // 0 or 8
  const _Float16* p0 = base + row * rowStride + kb + koff;
  v8h lo = *(const v8h*)p0;
  v8h hi = *(const v8h*)(p0 + 16);
  return combine_v8h(lo, hi);
}

// B-matrix 32x16 f16 fragment: lanes 0-15 hold column N=lane, K={0..7,16..23};
// lanes 16-31 hold column N=lane-16, K={8..15,24..31}.
// `base` must be laid out [N][K] contiguous in K.
static __device__ __forceinline__ v16h load_fragB(const _Float16* base,
                                                  int rowStride, int ntile,
                                                  int kb, int lane) {
  int n = (ntile << 4) + (lane & 15);
  int k = kb + ((lane >> 4) << 3);  // 0 or 8
  const _Float16* p = base + n * rowStride + k;
  v8h lo = *(const v8h*)p;
  v8h hi = *(const v8h*)(p + 16);
  return combine_v8h(lo, hi);
}

// ---------------------------------------------------------------------------
// CDNA5 async global->LDS (tracked by ASYNCcnt, data bypasses VGPRs).
// GVS addressing: 64-bit SGPR base + per-lane 32-bit byte offset.
// VDST VGPR carries the LDS byte address (low 32 bits of the generic pointer).
// ---------------------------------------------------------------------------
static __device__ __forceinline__ void async_b128_to_lds(unsigned lds_addr,
                                                         unsigned gbyte_off,
                                                         const void* gbase) {
  asm volatile("global_load_async_to_lds_b128 %0, %1, %2"
               :
               : "v"(lds_addr), "v"(gbyte_off), "s"(gbase)
               : "memory");
}

static __device__ __forceinline__ void wait_asynccnt0() {
#if __has_builtin(__builtin_amdgcn_s_wait_asynccnt)
  __builtin_amdgcn_s_wait_asynccnt(0);
#else
  asm volatile("s_wait_asynccnt 0x0" ::: "memory");
#endif
}

// ---------------------------------------------------------------------------
// Prep A: xT[b][n][c] f16 = transpose of x (pixel-major, feeds conv A-tiles).
// ---------------------------------------------------------------------------
__global__ void xprep_kernel(const float* __restrict__ x,
                             _Float16* __restrict__ xT) {
  int idx = (int)(blockIdx.x * blockDim.x + threadIdx.x);
  if (idx >= 8 * 4096 * 64) return;
  int c = idx & 63;
  int n = (idx >> 6) & 4095;
  int b = idx >> 18;
  xT[idx] = (_Float16)x[((size_t)b * 64 + c) * 4096 + n];
}

// ---------------------------------------------------------------------------
// Prep B: Wt[tap][o][c] f16 = Wstd[o][c][tap] + Laplacian folded on diagonal.
// lap taps (ky*3+kx): [0,1,0, 1,-4,1, 0,1,0]
// ---------------------------------------------------------------------------
__global__ void wprep_kernel(const float* __restrict__ Wstd,
                             _Float16* __restrict__ Wt) {
  int idx = (int)(blockIdx.x * blockDim.x + threadIdx.x);
  if (idx >= 9 * 64 * 64) return;
  int c = idx & 63;
  int o = (idx >> 6) & 63;
  int tap = idx >> 12;
  float v = Wstd[((size_t)o * 64 + c) * 9 + tap];
  if (o == c) v += (tap == 4) ? -4.0f : ((tap & 1) ? 1.0f : 0.0f);
  Wt[idx] = (_Float16)v;
}

// ---------------------------------------------------------------------------
// Kernel 1: implicit-GEMM 3x3 conv on the matrix pipe.
// Per (b, h): D[64 pixels x 64 out-ch] = sum_{9 taps} A_tap . B_tap with
// A_tap = shifted xT rows (pixel x in-ch), B_tap = Wt[tap] (o x c, read as
// [N][K]). Laplacian already folded into Wt. Activation rows streamed to LDS
// with async-to-LDS; border rows zero-filled (block-uniform branches only).
// Grid: (64 h, 8 b), block 128 = 4 waves; wave w owns pixel tile w*16..w*16+15.
// ---------------------------------------------------------------------------
__global__ __launch_bounds__(128) void conv_wmma_kernel(
    const _Float16* __restrict__ xT, const _Float16* __restrict__ Wt,
    float* __restrict__ feat, _Float16* __restrict__ fT1) {
  const int h = blockIdx.x;
  const int b = blockIdx.y;
  const int tid = (int)threadIdx.x;
  const int lane = tid & 31;
  const int ntile = tid >> 5;  // wave id = pixel tile

  constexpr int LS = 72;            // padded row stride (halves)
  __shared__ _Float16 xs[3 * 66 * LS];  // [dy][w+dx (0..65)][c]

  const _Float16* xb = xT + (size_t)b * 4096 * 64;
  const v8h zero8 = (v8h){};

  // Fill the three shifted row-planes.
  #pragma unroll
  for (int dy = 0; dy < 3; ++dy) {
    int hr = h + dy - 1;
    if (hr >= 0 && hr < 64) {
      // 64 rows x 128B each = 512 x 16B chunks, async global->LDS
      #pragma unroll
      for (int it = 0; it < 4; ++it) {
        int cid = tid + it * 128;
        int j = cid >> 3, c8 = (cid & 7) << 3;
        async_b128_to_lds((unsigned)(size_t)&xs[(dy * 66 + 1 + j) * LS + c8],
                          (unsigned)(((hr * 64 + j) * 64 + c8) * 2), xb);
      }
    } else {
      for (int idx = tid; idx < 512; idx += 128) {
        int j = idx >> 3, c8 = (idx & 7) << 3;
        *(v8h*)&xs[(dy * 66 + 1 + j) * LS + c8] = zero8;
      }
    }
  }
  // Zero the w=-1 / w=64 pad rows of each plane (6 rows x 8 chunks).
  for (int idx = tid; idx < 48; idx += 128) {
    int dy = idx >> 4, r = (idx >> 3) & 1, c8 = (idx & 7) << 3;
    *(v8h*)&xs[(dy * 66 + (r ? 65 : 0)) * LS + c8] = zero8;
  }
  wait_asynccnt0();
  __syncthreads();

  v8f oacc[4];
  #pragma unroll
  for (int i = 0; i < 4; ++i) oacc[i] = (v8f){};

  #pragma unroll
  for (int dy = 0; dy < 3; ++dy) {
    #pragma unroll
    for (int dx = 0; dx < 3; ++dx) {
      const int tap = dy * 3 + dx;
      const _Float16* abase = &xs[(dy * 66 + ntile * 16 + dx) * LS];
      v16h a0 = load_fragA(abase, LS, 0, lane);
      v16h a1 = load_fragA(abase, LS, 32, lane);
      const _Float16* wb = Wt + (size_t)tap * 64 * 64;  // [o][c]
      #pragma unroll
      for (int ct = 0; ct < 4; ++ct) {
        oacc[ct] = wmma16x16x32_f16(a0, load_fragB(wb, 64, ct, 0, lane), oacc[ct]);
        oacc[ct] = wmma16x16x32_f16(a1, load_fragB(wb, 64, ct, 32, lane), oacc[ct]);
      }
    }
  }

  // Store feat (f32, NCHW) and fT1 (f16, pixel-major) from D accumulators.
  #pragma unroll
  for (int r = 0; r < 8; ++r) {
    int w = ntile * 16 + ((lane < 16) ? r : r + 8);
    #pragma unroll
    for (int ct = 0; ct < 4; ++ct) {
      int o = (ct << 4) + (lane & 15);
      float val = oacc[ct][r];
      feat[(((size_t)b * 64 + o) * 64 + h) * 64 + w] = val;
      fT1[(((size_t)b * 4096) + h * 64 + w) * 64 + o] = (_Float16)val;
    }
  }
}

// ---------------------------------------------------------------------------
// Kernel 2: exact-divisor average pool -> f16 transposed features fT[b][n][c]
// ---------------------------------------------------------------------------
__global__ void pool_kernel(const float* __restrict__ feat,
                            _Float16* __restrict__ fT, int S, int HP) {
  int total = 8 * HP * HP * 64;
  int idx = (int)(blockIdx.x * blockDim.x + threadIdx.x);
  if (idx >= total) return;
  int c = idx & 63;
  int n = (idx >> 6) % (HP * HP);
  int b = idx / (64 * HP * HP);
  int h2 = n / HP, w2 = n % HP;
  float s = 0.0f;
  for (int dy = 0; dy < S; ++dy)
    for (int dx = 0; dx < S; ++dx)
      s += feat[(((size_t)b * 64 + c) * 64 + h2 * S + dy) * 64 + w2 * S + dx];
  fT[idx] = (_Float16)(s / (float)(S * S));
}

// ---------------------------------------------------------------------------
// Kernel 3: flash non-local attention. fT is [B][N][64] f16 (Q=K=V).
// O is [B][64][N] f32: O[b,c,n] = sum_m softmax_m(f_n . f_m) * f[c,m].
// Block 128 = 4 waves; each wave owns 16 query rows of a 64-row query tile.
// KV tiles are streamed global->LDS with async-to-LDS loads, double buffered.
// Grid: (N/64, B).
// ---------------------------------------------------------------------------
template <int N>
__global__ __launch_bounds__(128) void attn_kernel(
    const _Float16* __restrict__ fT, float* __restrict__ O) {
  const int qt = blockIdx.x;
  const int b = blockIdx.y;
  const int tid = (int)threadIdx.x;
  const int lane = tid & 31;
  const int wq = (tid >> 5) << 4;  // this wave's query-row offset in the tile

  constexpr int LS = 72;  // LDS row stride in halves (64 + 8 pad, 16B aligned)
  __shared__ _Float16 Ks[2][64 * LS];  // double-buffered KV tile, [kv_row][ch]
  __shared__ _Float16 Kt[64 * LS];     // KV tile transposed, [ch][kv_row]
  __shared__ _Float16 Ps[64 * LS];     // probabilities, [q_row][kv_col]

  const _Float16* fbase = fT + (size_t)b * N * 64;

  // Q fragments: loaded once, straight from global (fT rows are [n][c]).
  const _Float16* qbase = fbase + (size_t)(qt * 64 + wq) * 64;
  v16h aq0 = load_fragA(qbase, 64, 0, lane);
  v16h aq1 = load_fragA(qbase, 64, 32, lane);

  v8f oacc[4];
  float rM[8], rL[8];
  #pragma unroll
  for (int i = 0; i < 4; ++i) oacc[i] = (v8f){};
  #pragma unroll
  for (int r = 0; r < 8; ++r) { rM[r] = -3.0e38f; rL[r] = 0.0f; }

  // Issue async fill of KV tile 0 into buffer 0 (4 x b128 per thread).
  #pragma unroll
  for (int it = 0; it < 4; ++it) {
    int cid = tid + it * 128;  // 0..511 : 16B chunk id of the 8KB tile
    int j = cid >> 3, c8 = (cid & 7) << 3;
    async_b128_to_lds((unsigned)(size_t)&Ks[0][j * LS + c8],
                      (unsigned)(cid << 4), fbase);
  }

  for (int t = 0; t < N / 64; ++t) {
    const int cur = t & 1;
    wait_asynccnt0();   // our async chunks of tile t have landed in LDS
    __syncthreads();    // everyone's chunks landed; all readers of buf cur^1 done

    // Prefetch tile t+1 into the other buffer while we compute on tile t.
    if (t + 1 < N / 64) {
      const char* nsrc = (const char*)(fbase + (size_t)(t + 1) * 64 * 64);
      #pragma unroll
      for (int it = 0; it < 4; ++it) {
        int cid = tid + it * 128;
        int j = cid >> 3, c8 = (cid & 7) << 3;
        async_b128_to_lds((unsigned)(size_t)&Ks[cur ^ 1][j * LS + c8],
                          (unsigned)(cid << 4), nsrc);
      }
    }

    // Build the transposed copy (feeds B-fragments of O += P.V).
    for (int cid = tid; cid < 512; cid += 128) {
      int j = cid >> 3, c8 = (cid & 7) << 3;
      v8h v = *(const v8h*)&Ks[cur][j * LS + c8];
      #pragma unroll
      for (int i = 0; i < 8; ++i) Kt[(c8 + i) * LS + j] = v[i];
    }
    __syncthreads();

    // S = Q . K^T  (16 q-rows x 64 kv-cols, inner dim C=64)
    v8f sacc[4];
    #pragma unroll
    for (int mt = 0; mt < 4; ++mt) {
      v8f acc = (v8f){};
      acc = wmma16x16x32_f16(aq0, load_fragB(Ks[cur], LS, mt, 0, lane), acc);
      acc = wmma16x16x32_f16(aq1, load_fragB(Ks[cur], LS, mt, 32, lane), acc);
      sacc[mt] = acc;
    }

    // Online softmax per accumulator row r (M=r in lanes 0-15, M=r+8 above).
    #pragma unroll
    for (int r = 0; r < 8; ++r) {
      float vmax = fmaxf(fmaxf(sacc[0][r], sacc[1][r]),
                         fmaxf(sacc[2][r], sacc[3][r]));
      #pragma unroll
      for (int m = 1; m < 16; m <<= 1) vmax = fmaxf(vmax, __shfl_xor(vmax, m, 32));
      float newM = fmaxf(rM[r], vmax);
      float corr = __expf(rM[r] - newM);
      rM[r] = newM;
      int row = wq + ((lane < 16) ? r : r + 8);
      float psum = 0.0f;
      #pragma unroll
      for (int mt = 0; mt < 4; ++mt) {
        float p = __expf(sacc[mt][r] - newM);
        psum += p;
        Ps[row * LS + (mt << 4) + (lane & 15)] = (_Float16)p;
      }
      #pragma unroll
      for (int m = 1; m < 16; m <<= 1) psum += __shfl_xor(psum, m, 32);
      rL[r] = rL[r] * corr + psum;
      #pragma unroll
      for (int ct = 0; ct < 4; ++ct) oacc[ct][r] *= corr;
    }

    // O += P . V   (V rows are the same KV tile; B-frags from transposed copy)
    v16h ap0 = load_fragA(&Ps[wq * LS], LS, 0, lane);
    v16h ap1 = load_fragA(&Ps[wq * LS], LS, 32, lane);
    #pragma unroll
    for (int ct = 0; ct < 4; ++ct) {
      oacc[ct] = wmma16x16x32_f16(ap0, load_fragB(Kt, LS, ct, 0, lane), oacc[ct]);
      oacc[ct] = wmma16x16x32_f16(ap1, load_fragB(Kt, LS, ct, 32, lane), oacc[ct]);
    }
  }

  // Finalize (divide by row sums) and store O[b][c][n].
  #pragma unroll
  for (int r = 0; r < 8; ++r) {
    float inv = 1.0f / rL[r];
    int n = qt * 64 + wq + ((lane < 16) ? r : r + 8);
    #pragma unroll
    for (int ct = 0; ct < 4; ++ct) {
      int c = (ct << 4) + (lane & 15);
      O[((size_t)b * 64 + c) * N + n] = oacc[ct][r] * inv;
    }
  }
}

// ---------------------------------------------------------------------------
// Kernel 4: out = x + O1 + upsample(O2) + upsample(O4); half-pixel bilinear
// with clamped indices (== torch align_corners=False for upsampling).
// ---------------------------------------------------------------------------
static __device__ __forceinline__ float bilerp(const float* __restrict__ O,
                                               int b, int c, int P,
                                               float sh, float sw) {
  float fh = floorf(sh), fw = floorf(sw);
  int h0 = (int)fh, w0 = (int)fw;
  float ah = sh - fh, aw = sw - fw;
  int h1 = h0 + 1, w1 = w0 + 1;
  h0 = min(max(h0, 0), P - 1); h1 = min(max(h1, 0), P - 1);
  w0 = min(max(w0, 0), P - 1); w1 = min(max(w1, 0), P - 1);
  const float* base = O + ((size_t)b * 64 + c) * (P * P);
  float v00 = base[h0 * P + w0], v01 = base[h0 * P + w1];
  float v10 = base[h1 * P + w0], v11 = base[h1 * P + w1];
  return (1.0f - ah) * ((1.0f - aw) * v00 + aw * v01)
       + ah * ((1.0f - aw) * v10 + aw * v11);
}

__global__ void combine_kernel(const float* __restrict__ x,
                               const float* __restrict__ O1,
                               const float* __restrict__ O2,
                               const float* __restrict__ O4,
                               float* __restrict__ out) {
  int idx = (int)(blockIdx.x * blockDim.x + threadIdx.x);
  if (idx >= 8 * 64 * 64 * 64) return;
  int w = idx & 63;
  int h = (idx >> 6) & 63;
  int c = (idx >> 12) & 63;
  int b = idx >> 18;
  float v = x[idx] + O1[((size_t)b * 64 + c) * 4096 + h * 64 + w];
  v += bilerp(O2, b, c, 32, (h + 0.5f) * 0.5f - 0.5f, (w + 0.5f) * 0.5f - 0.5f);
  v += bilerp(O4, b, c, 16, (h + 0.5f) * 0.25f - 0.5f, (w + 0.5f) * 0.25f - 0.5f);
  out[idx] = v;
}

// ---------------------------------------------------------------------------
extern "C" void kernel_launch(void* const* d_in, const int* in_sizes, int n_in,
                              void* d_out, int out_size, void* d_ws, size_t ws_size,
                              hipStream_t stream) {
  const float* x    = (const float*)d_in[0];  // [8,64,64,64] f32
  const float* Wstd = (const float*)d_in[1];  // [64,64,3,3] f32
  float* out = (float*)d_out;                 // [8,64,64,64] f32

  char* ws = (char*)d_ws;
  float*    feat = (float*)   (ws + 0);         // [8,64,64,64] f32, 8 MB
  _Float16* xT   = (_Float16*)(ws + 8388608);   // [8,4096,64] f16, 4 MB
  _Float16* Wt   = (_Float16*)(ws + 12582912);  // [9,64,64]   f16, 72 KB
  _Float16* fT1  = (_Float16*)(ws + 12656640);  // [8,4096,64] f16, 4 MB
  _Float16* fT2  = (_Float16*)(ws + 16850944);  // [8,1024,64] f16, 1 MB
  _Float16* fT4  = (_Float16*)(ws + 17899520);  // [8, 256,64] f16, 256 KB
  float*    O1   = (float*)   (ws + 18161664);  // [8,64,4096] f32, 8 MB
  float*    O2   = (float*)   (ws + 26550272);  // [8,64,1024] f32, 2 MB
  float*    O4   = (float*)   (ws + 28647424);  // [8,64, 256] f32, 512 KB

  xprep_kernel<<<(8 * 4096 * 64 + 255) / 256, 256, 0, stream>>>(x, xT);
  wprep_kernel<<<(9 * 64 * 64 + 255) / 256, 256, 0, stream>>>(Wstd, Wt);
  conv_wmma_kernel<<<dim3(64, 8), 128, 0, stream>>>(xT, Wt, feat, fT1);

  pool_kernel<<<(8 * 32 * 32 * 64 + 255) / 256, 256, 0, stream>>>(feat, fT2, 2, 32);
  pool_kernel<<<(8 * 16 * 16 * 64 + 255) / 256, 256, 0, stream>>>(feat, fT4, 4, 16);

  attn_kernel<4096><<<dim3(64, 8), 128, 0, stream>>>(fT1, O1);
  attn_kernel<1024><<<dim3(16, 8), 128, 0, stream>>>(fT2, O2);
  attn_kernel<256> <<<dim3(4, 8),  128, 0, stream>>>(fT4, O4);

  combine_kernel<<<(8 * 64 * 64 * 64 + 255) / 256, 256, 0, stream>>>(x, O1, O2, O4, out);
}
